// OntologyGNN_47150150975760
// MI455X (gfx1250) — matrix operand order repaired and checked by reading, not compiled, not run on hardware
//
#include <hip/hip_runtime.h>
#include <hip/hip_bf16.h>

// ---------------------------------------------------------------------------
// 2-layer GCN for MI455X (gfx1250, wave32, WMMA).
//   Ab = bf16 row-major copy of GEMM input (once per layer; L2-resident)
//   Wt = bf16 transposed copy of W        (once; L2-resident)
//   GEMM k-loop: 8x global_load_b128 frags + 4x v_wmma, no LDS, no barriers
//   agg[i]   = h[i]*dinv[i]^2 + b                  (self loop + bias)
//   agg[dst] += h[src]*dinv[src]*dinv[dst]         (f32 atomics, L2-resident)
// ---------------------------------------------------------------------------

typedef __attribute__((ext_vector_type(16))) __bf16 bf16x16;
typedef __attribute__((ext_vector_type(8)))  float  floatx8;

static __device__ __forceinline__ unsigned int bfhi(float f) {
    return (__float_as_uint(f) + 0x8000u) >> 16;       // round-half-up bf16
}
static __device__ __forceinline__ unsigned int pack2bf(float lo, float hi) {
    return bfhi(lo) | (bfhi(hi) << 16);
}

static __device__ __forceinline__ bf16x16 load_frag(const unsigned short* p,
                                                    int off1) {
    union { bf16x16 v; uint4 q[2]; } u;
    u.q[0] = *reinterpret_cast<const uint4*>(p);
    u.q[1] = *reinterpret_cast<const uint4*>(p + off1);
    return u.v;
}

// -------------------------- degree / normalization -------------------------

__global__ void gcn_deg_init(float* deg, int n) {
    int i = blockIdx.x * blockDim.x + threadIdx.x;
    if (i < n) deg[i] = 1.0f;                    // self loop contributes 1
}

__global__ void gcn_deg_scatter(const int* dst, float* deg, int e) {
    int i = blockIdx.x * blockDim.x + threadIdx.x;
    if (i < e) atomicAdd(&deg[dst[i]], 1.0f);
}

__global__ void gcn_dinv(const float* deg, float* dinv, int n) {
    int i = blockIdx.x * blockDim.x + threadIdx.x;
    if (i < n) dinv[i] = rsqrtf(deg[i]);         // deg >= 1 always
}

// W [K][N] fp32 -> Wt [N][K] bf16 (coalesced read along N)
__global__ void gcn_wcvt(const float* __restrict__ W,
                         unsigned short* __restrict__ Wt, int K, int N) {
    int idx = blockIdx.x * blockDim.x + threadIdx.x;
    if (idx >= K * N) return;
    int k = idx / N, n = idx - k * N;
    Wt[n * K + k] = (unsigned short)bfhi(W[idx]);
}

// fp32 -> packed bf16, streaming (float4 -> uint2); optional fused ReLU
template <int RELU>
__global__ void gcn_cvt_bf16(const float* __restrict__ in,
                             unsigned short* __restrict__ out, int total4) {
    int idx = blockIdx.x * blockDim.x + threadIdx.x;
    if (idx >= total4) return;
    float4 v = reinterpret_cast<const float4*>(in)[idx];
    if (RELU) {
        v.x = fmaxf(v.x, 0.f); v.y = fmaxf(v.y, 0.f);
        v.z = fmaxf(v.z, 0.f); v.w = fmaxf(v.w, 0.f);
    }
    reinterpret_cast<uint2*>(out)[idx] =
        make_uint2(pack2bf(v.x, v.y), pack2bf(v.z, v.w));
}

// ------------------------------- WMMA GEMM ---------------------------------
// C[M,N] = Ab[M,K](bf16) @ Wt[N][K](bf16). Ab rows padded to TM multiple.
// Block tile 128x64, 8 waves (wave32), each wave a 32x32 sub-tile (2x2 WMMA).
// No LDS / barriers in the k-loop: all fragments are global_load_b128 pairs
// from L2-resident bf16 buffers.

#define TM 128
#define TN 64
#define TK 32

__global__ __launch_bounds__(256)
void gcn_gemm_wmma(const unsigned short* __restrict__ Ab,
                   const unsigned short* __restrict__ Wt,
                   float* __restrict__ C, int M, int K, int N) {
    __shared__ __align__(16) float sC[TM * TN];  // epilogue staging (32KB)

    const int tid   = threadIdx.x;
    const int lane  = tid & 31;
    const int wave  = tid >> 5;        // 0..7
    const int wm    = wave & 3;        // M sub-tile (32 rows each)
    const int wn    = wave >> 2;       // N sub-tile (32 cols each)
    const int row16 = lane & 15;
    const int hi    = lane >> 4;       // half-wave select

    const int m0 = blockIdx.x * TM;
    const int n0 = blockIdx.y * TN;

    // Per-lane fragment base pointers (row-major bf16, K contiguous)
    const unsigned short* pa[2];
    const unsigned short* pb[2];
    #pragma unroll
    for (int mf = 0; mf < 2; ++mf)
        pa[mf] = Ab + (unsigned)(m0 + wm * 32 + mf * 16 + row16) * (unsigned)K
                    + hi * 8;          // A: halfs 0..7=K klo..+7, 8..15=+16..+23
    #pragma unroll
    for (int nf = 0; nf < 2; ++nf)
        pb[nf] = Wt + (unsigned)(n0 + wn * 32 + nf * 16 + row16) * (unsigned)K
                    + hi * 8;          // B: halfs 0..7=K klo..+7, 8..15=+16..+23

    floatx8 acc[2][2];
    acc[0][0] = (floatx8)0.0f; acc[0][1] = (floatx8)0.0f;
    acc[1][0] = (floatx8)0.0f; acc[1][1] = (floatx8)0.0f;

    #pragma unroll 2
    for (int kt = 0; kt < K; kt += TK) {
        bf16x16 aF[2], bF[2];
        #pragma unroll
        for (int mf = 0; mf < 2; ++mf) aF[mf] = load_frag(pa[mf] + kt, 16);
        #pragma unroll
        for (int nf = 0; nf < 2; ++nf) bF[nf] = load_frag(pb[nf] + kt, 16);
        #pragma unroll
        for (int mf = 0; mf < 2; ++mf)
            #pragma unroll
            for (int nf = 0; nf < 2; ++nf)
                acc[mf][nf] = __builtin_amdgcn_wmma_f32_16x16x32_bf16(
                    false, aF[mf], false, bF[nf], (short)0, acc[mf][nf],
                    false, false);
    }

    // Epilogue: stage to LDS, then coalesced float4 stores (M-tail guarded)
    #pragma unroll
    for (int mf = 0; mf < 2; ++mf)
        #pragma unroll
        for (int nf = 0; nf < 2; ++nf) {
            int mr = wm * 32 + mf * 16 + hi * 8;
            int nc = wn * 32 + nf * 16 + row16;
            #pragma unroll
            for (int v = 0; v < 8; ++v)
                sC[(mr + v) * TN + nc] = acc[mf][nf][v];
        }
    __syncthreads();

    #pragma unroll
    for (int it = 0; it < 8; ++it) {
        int j  = tid + it * 256;       // over TM*TN/4 float4s
        int m  = j >> 4;               // 16 float4s per row
        int nq = j & 15;
        int gm = m0 + m;
        if (gm < M) {
            *reinterpret_cast<float4*>(C + (unsigned)gm * (unsigned)N + n0 + nq * 4) =
                *reinterpret_cast<const float4*>(sC + m * TN + nq * 4);
        }
    }
}

// ----------------------- aggregation (self loop + edges) -------------------

// out[i][c] = h[i][c] * dinv[i]^2 + b[c]   (float4 per thread, overwrites out)
__global__ void gcn_agg_init(const float* __restrict__ h,
                             const float* __restrict__ dinv,
                             const float* __restrict__ b,
                             float* __restrict__ out, int n) {
    int idx = blockIdx.x * blockDim.x + threadIdx.x;   // over n*64 float4s
    if (idx >= n * 64) return;
    int i = idx >> 6;
    int c = (idx & 63) * 4;
    float w = dinv[i] * dinv[i];
    const float4 hv = *reinterpret_cast<const float4*>(h + (unsigned)i * 256u + c);
    const float4 bv = *reinterpret_cast<const float4*>(b + c);
    float4 o;
    o.x = hv.x * w + bv.x; o.y = hv.y * w + bv.y;
    o.z = hv.z * w + bv.z; o.w = hv.w * w + bv.w;
    *reinterpret_cast<float4*>(out + (unsigned)i * 256u + c) = o;
}

// out[dst] += h[src] * dinv[src]*dinv[dst]; 64 lanes x float4 per edge,
// 4 edges per block. h/out are L2-resident (51 MB each << 192 MB L2).
__global__ void gcn_agg_scatter(const float* __restrict__ h,
                                const int* __restrict__ src,
                                const int* __restrict__ dst,
                                const float* __restrict__ dinv,
                                float* __restrict__ out, int e) {
    int eidx = blockIdx.x * 4 + (threadIdx.x >> 6);
    if (eidx >= e) return;
    int c = (threadIdx.x & 63) * 4;
    int s = src[eidx], d = dst[eidx];
    float w = dinv[s] * dinv[d];
    const float4 v = *reinterpret_cast<const float4*>(h + (unsigned)s * 256u + c);
    float* o = out + (unsigned)d * 256u + c;
    atomicAdd(o + 0, v.x * w);
    atomicAdd(o + 1, v.y * w);
    atomicAdd(o + 2, v.z * w);
    atomicAdd(o + 3, v.w * w);
}

// ------------------------------- launcher ----------------------------------

extern "C" void kernel_launch(void* const* d_in, const int* in_sizes, int n_in,
                              void* d_out, int out_size, void* d_ws, size_t ws_size,
                              hipStream_t stream) {
    const float* x  = (const float*)d_in[0];
    const int*   ei = (const int*)  d_in[1];
    const float* W1 = (const float*)d_in[2];
    const float* b1 = (const float*)d_in[3];
    const float* W2 = (const float*)d_in[4];
    const float* b2 = (const float*)d_in[5];

    const int N  = in_sizes[0] / 768;   // 50000 nodes
    const int E  = in_sizes[1] / 2;     // 800000 edges
    const int C  = 256;
    const int K1 = 768;
    const int Mp = ((N + TM - 1) / TM) * TM;   // rows padded to 128 (50048)

    const int* src = ei;
    const int* dst = ei + E;

    float* deg  = (float*)d_ws;
    float* dinv = deg  + N;
    float* h    = dinv + N;                         // [N, 256] fp32
    float* agg  = h + (size_t)N * C;                // [N, 256] fp32
    unsigned short* wt1 = (unsigned short*)(agg + (size_t)N * C); // [256][768]
    unsigned short* wt2 = wt1 + (size_t)K1 * C;                   // [256][256]
    unsigned short* ab  = wt2 + (size_t)C * C;      // [Mp, 768] bf16 (reused)
    float* out  = (float*)d_out;                    // [N, 256] fp32

    // Weight pre-conversion (bf16, transposed) + normalization
    gcn_wcvt<<<(K1 * C + 255) / 256, 256, 0, stream>>>(W1, wt1, K1, C);
    gcn_wcvt<<<(C * C + 255) / 256, 256, 0, stream>>>(W2, wt2, C, C);
    gcn_deg_init   <<<(N + 255) / 256, 256, 0, stream>>>(deg, N);
    gcn_deg_scatter<<<(E + 255) / 256, 256, 0, stream>>>(dst, deg, E);
    gcn_dinv       <<<(N + 255) / 256, 256, 0, stream>>>(deg, dinv, N);

    dim3 gemmGrid(Mp / TM, C / TN);                 // (391, 4)

    // Layer 1: h = x @ W1 ; agg = norm-aggregate(h) + b1
    int t41 = N * K1 / 4;
    gcn_cvt_bf16<0><<<(t41 + 255) / 256, 256, 0, stream>>>(x, ab, t41);
    gcn_gemm_wmma<<<gemmGrid, 256, 0, stream>>>(ab, wt1, h, N, K1, C);
    gcn_agg_init <<<(N * 64 + 255) / 256, 256, 0, stream>>>(h, dinv, b1, agg, N);
    gcn_agg_scatter<<<(E + 3) / 4, 256, 0, stream>>>(h, src, dst, dinv, agg, E);

    // Layer 2: h = relu(agg) @ W2 ; out = norm-aggregate(h) + b2
    int t42 = N * C / 4;
    gcn_cvt_bf16<1><<<(t42 + 255) / 256, 256, 0, stream>>>(agg, ab, t42);
    gcn_gemm_wmma<<<gemmGrid, 256, 0, stream>>>(ab, wt2, h, N, C, C);
    gcn_agg_init <<<(N * 64 + 255) / 256, 256, 0, stream>>>(h, dinv, b2, out, N);
    gcn_agg_scatter<<<(E + 3) / 4, 256, 0, stream>>>(h, src, dst, dinv, out, E);
}
